// ImageGNN_45801531245237
// MI455X (gfx1250) — matrix operand order repaired and checked by reading, not compiled, not run on hardware
//
#include <hip/hip_runtime.h>
#include <hip/hip_bf16.h>

typedef __attribute__((ext_vector_type(16))) _Float16 v16h;
typedef __attribute__((ext_vector_type(8)))  _Float16 v8h;
typedef __attribute__((ext_vector_type(8)))  float    v8f;
typedef __attribute__((ext_vector_type(4)))  float    vf4;

union AFrag { v16h v; vf4 q[2]; _Float16 h[16]; };

// A-fragment from an LDS row of concatenated features.
// Lane l: M = l%16, g = l/16. Halves j=0..7 -> K = kbase+j ; j=8..15 -> K = kbase+16+(j-8)
// where kbase = kt*32 + 8*g  (matches the ISA 16-bit A 16x32 layout).
__device__ inline v16h ldsFrag(const _Float16* row, int kbase) {
  AFrag f;
  f.q[0] = *(const vf4*)(row + kbase);
  f.q[1] = *(const vf4*)(row + kbase + 16);
  return f.v;
}

// B-fragment from pre-packed weights: 32 lanes x 16 contiguous halves per fragment.
__device__ inline v16h glbFrag(const _Float16* p) {
  AFrag f;
  f.q[0] = *(const vf4*)(p);
  f.q[1] = *(const vf4*)(p + 8);
  return f.v;
}

__device__ inline v8f wmma32(v16h a, v16h b, v8f c) {
  return __builtin_amdgcn_wmma_f32_16x16x32_f16(false, a, false, b, (short)0, c, false, false);
}

// ---------------------------------------------------------------------------
// Pack f32 weights into f16 B-fragment layout.
// Fragment f = kt*NT + nt. Element (f, lane l, half j):
//   K = kt*32 + 8*(l/16) + j (j<8), kt*32 + 16 + 8*(l/16) + (j-8) (j>=8) ; N = nt*16 + (l%16)
//   value = K<F ? wn[K,N] : (K<2F && wr) ? wr[K-F,N] : 0   (N >= Ncols -> 0)
// ---------------------------------------------------------------------------
__global__ void pack_w(const float* __restrict__ w0, const float* __restrict__ w1,
                       int F, int Ncols, int NT, int KT, int total,
                       _Float16* __restrict__ dst) {
  int idx = blockIdx.x * 256 + threadIdx.x;
  if (idx >= total) return;
  int j  = idx & 15;
  int l  = (idx >> 4) & 31;
  int f  = idx >> 9;
  int nt = f % NT;
  int kt = f / NT;
  int K = kt * 32 + (j < 8 ? 8 * (l >> 4) + j : 16 + 8 * (l >> 4) + (j - 8));
  int N = nt * 16 + (l & 15);
  float v = 0.f;
  if (N < Ncols) {
    if (K < F)                        v = w0[K * Ncols + N];
    else if (w1 != nullptr && K < 2*F) v = w1[(K - F) * Ncols + N];
  }
  dst[idx] = (_Float16)v;
}

// ---------------------------------------------------------------------------
// Patchify: image [4,512,512] f32 -> X1h [B*128*128, 16] f16 (node-major).
// ---------------------------------------------------------------------------
__global__ void patchify1(const float* __restrict__ img, _Float16* __restrict__ x1h) {
  int node = blockIdx.x * blockDim.x + threadIdx.x;           // 65536 nodes
  int b = node >> 14;
  int i = (node >> 7) & 127;
  int j = node & 127;
  const vf4* p = (const vf4*)(img + ((size_t)(b * 512 + i * 4)) * 512 + j * 4);
  union { _Float16 h[16]; vf4 q[2]; } o;
#pragma unroll
  for (int r = 0; r < 4; ++r) {
    vf4 v = p[(size_t)r * 128];
    o.h[r * 4 + 0] = (_Float16)v.x;
    o.h[r * 4 + 1] = (_Float16)v.y;
    o.h[r * 4 + 2] = (_Float16)v.z;
    o.h[r * 4 + 3] = (_Float16)v.w;
  }
  vf4* d = (vf4*)(x1h + (size_t)node * 16);
  d[0] = o.q[0];
  d[1] = o.q[1];
}

// ---------------------------------------------------------------------------
// First GraphConv of a level: out = [x|agg] @ Wcat + b,  F in {16, 8}, Kpad=32.
// One wave handles 16 consecutive nodes of one grid row. 4 WMMA per wave.
// ---------------------------------------------------------------------------
template<int F>
__global__ __launch_bounds__(128)
void conv_first(const _Float16* __restrict__ xin, const _Float16* __restrict__ wpk,
                const float* __restrict__ bias, _Float16* __restrict__ out, int n) {
  const int lane = threadIdx.x & 31;
  const int w    = threadIdx.x >> 5;
  const int tpr  = n >> 4;
  const int tile = blockIdx.x * 4 + w;
  const int jt = tile % tpr;
  const int i  = (tile / tpr) % n;
  const int b  = tile / (tpr * n);
  const long row0 = (long)(b * n + i) * n;

  __shared__ _Float16 sA[4][16][32];
  constexpr int C = F / 8;                       // v8h chunks per feature row

  if (lane < 16) {
    const int j = jt * 16 + lane;
    const long node = row0 + j;
    const v8h* xc = (const v8h*)(xin + (size_t)node * F);
    v8h* dX = (v8h*)&sA[w][lane][0];
    v8h* dS = (v8h*)&sA[w][lane][F];
    v8h  z  = {};
#pragma unroll
    for (int k = 0; k < C; ++k) {
      v8h x = xc[k];
      v8h s = z;
      if (j > 0)     s += xc[k - C];
      if (j < n - 1) s += xc[k + C];
      if (i > 0)     s += ((const v8h*)(xin + (size_t)(node - n) * F))[k];
      if (i < n - 1) s += ((const v8h*)(xin + (size_t)(node + n) * F))[k];
      dX[k] = x;
      dS[k] = s;
    }
#pragma unroll
    for (int k = 2 * C; k < 4; ++k) ((v8h*)&sA[w][lane][0])[k] = z;   // zero pad K to 32
  }
  __syncthreads();

  const int g = lane >> 4, m = lane & 15;
  v16h a = ldsFrag(&sA[w][m][0], 8 * g);
#pragma unroll
  for (int nt = 0; nt < 4; ++nt) {
    v8f acc = {};
    acc = wmma32(a, glbFrag(wpk + ((size_t)nt * 32 + lane) * 16), acc);
    int ncol = nt * 16 + m;
    float bb = bias[ncol];
#pragma unroll
    for (int r = 0; r < 8; ++r) {
      long node = row0 + jt * 16 + r + 8 * g;                 // M = r + 8g
      out[(size_t)node * 64 + ncol] = (_Float16)(acc[r] + bb);
    }
  }
}

// ---------------------------------------------------------------------------
// Fused: conv2 (K=128, 16 WMMA) + relu + proj (2 WMMA) + sol + epilogue.
// LEVEL==1: writes sol1 and builds level-2 node features X2h.
// LEVEL==2: writes sol2, final_img, final_cs.
// ---------------------------------------------------------------------------
template<int LEVEL>
__global__ __launch_bounds__(128)
void conv_second(const _Float16* __restrict__ hin,
                 const _Float16* __restrict__ wcat, const float* __restrict__ bias,
                 const _Float16* __restrict__ wproj, const float* __restrict__ projb,
                 const float* __restrict__ solw, const float* __restrict__ solb,
                 float* __restrict__ out_sol, _Float16* __restrict__ x2h,
                 float* __restrict__ out_img, float* __restrict__ out_cs, int n) {
  const int lane = threadIdx.x & 31;
  const int w    = threadIdx.x >> 5;
  const int tpr  = n >> 4;
  const int tile = blockIdx.x * 4 + w;
  const int jt = tile % tpr;
  const int i  = (tile / tpr) % n;
  const int b  = tile / (tpr * n);
  const long row0 = (long)(b * n + i) * n;

  __shared__ _Float16 sA[4][16][128];
  __shared__ _Float16 sH[4][16][64];
  __shared__ float    sY[4][16][16];

  if (lane < 16) {                               // stage [x(64) | agg(64)] per node
    const int j = jt * 16 + lane;
    const long node = row0 + j;
    const v8h* xc = (const v8h*)(hin + (size_t)node * 64);
    v8h* dX = (v8h*)&sA[w][lane][0];
    v8h* dS = (v8h*)&sA[w][lane][64];
    v8h  z  = {};
#pragma unroll
    for (int k = 0; k < 8; ++k) {
      v8h x = xc[k];
      v8h s = z;
      if (j > 0)     s += xc[k - 8];
      if (j < n - 1) s += xc[k + 8];
      if (i > 0)     s += ((const v8h*)(hin + (size_t)(node - n) * 64))[k];
      if (i < n - 1) s += ((const v8h*)(hin + (size_t)(node + n) * 64))[k];
      dX[k] = x;
      dS[k] = s;
    }
  }
  __syncthreads();

  const int g = lane >> 4, m = lane & 15;
  v8f acc[4] = {{}, {}, {}, {}};
#pragma unroll
  for (int kt = 0; kt < 4; ++kt) {
    v16h a = ldsFrag(&sA[w][m][0], kt * 32 + 8 * g);
#pragma unroll
    for (int nt = 0; nt < 4; ++nt)
      acc[nt] = wmma32(a, glbFrag(wcat + ((size_t)(kt * 4 + nt) * 32 + lane) * 16), acc[nt]);
  }
  // bias + relu -> f16, restaged in LDS as A-operand for the projection GEMM
#pragma unroll
  for (int nt = 0; nt < 4; ++nt) {
    int ncol = nt * 16 + m;
    float bb = bias[ncol];
#pragma unroll
    for (int r = 0; r < 8; ++r) {
      float v = acc[nt][r] + bb;
      v = v > 0.f ? v : 0.f;
      sH[w][r + 8 * g][ncol] = (_Float16)v;
    }
  }
  __syncthreads();

  v8f pacc = {};
#pragma unroll
  for (int kt = 0; kt < 2; ++kt) {
    v16h a = ldsFrag(&sH[w][m][0], kt * 32 + 8 * g);
    pacc = wmma32(a, glbFrag(wproj + ((size_t)kt * 32 + lane) * 16), pacc);
  }
#pragma unroll
  for (int r = 0; r < 8; ++r) sY[w][r + 8 * g][m] = pacc[r];
  __syncthreads();

  if (lane < 16) {                               // per-node scalar epilogue
    const int j = jt * 16 + lane;
    const long node = row0 + j;
    if (LEVEL == 1) {
      float y[16];
      float sol = solb[0];
#pragma unroll
      for (int k = 0; k < 16; ++k) {
        y[k] = sY[w][lane][k] + projb[k];
        sol += y[k] * solw[k];
      }
      sol = fminf(fmaxf(sol, -4.f), 4.f);
      out_sol[node] = sol;
      // spawn 4 level-2 children: [patchify(img1,2) | patchify(cs1,2)]
#pragma unroll
      for (int si = 0; si < 2; ++si)
#pragma unroll
        for (int sj = 0; sj < 2; ++sj) {
          int qi = 2 * i + si, qj = 2 * j + sj;
          size_t node2 = ((size_t)(b * 256 + qi)) * 256 + qj;
          union { _Float16 h[8]; vf4 q; } cc;
#pragma unroll
          for (int r2 = 0; r2 < 2; ++r2)
#pragma unroll
            for (int c2 = 0; c2 < 2; ++c2) {
              int k2 = r2 * 2 + c2;
              float v = (y[(2 * si + r2) * 4 + (2 * sj + c2)] + 4.f) * (255.f / 8.f);
              cc.h[k2]     = (_Float16)v;
              cc.h[4 + k2] = (_Float16)sol;
            }
          *((vf4*)(x2h + node2 * 8)) = cc.q;
        }
    } else {
      float y[4];
      float sol = solb[0];
#pragma unroll
      for (int k = 0; k < 4; ++k) {
        y[k] = sY[w][lane][k] + projb[k];
        sol += y[k] * solw[k];
      }
      sol = fminf(fmaxf(sol, -4.f), 4.f);
      out_sol[node] = sol;
#pragma unroll
      for (int r = 0; r < 2; ++r)
#pragma unroll
        for (int c = 0; c < 2; ++c) {
          size_t px = ((size_t)b * 512 + (2 * i + r)) * 512 + (2 * j + c);
          out_img[px] = (y[r * 2 + c] + 4.f) * (255.f / 8.f);
          out_cs[px]  = sol;
        }
    }
  }
}

// ---------------------------------------------------------------------------
extern "C" void kernel_launch(void* const* d_in, const int* in_sizes, int n_in,
                              void* d_out, int out_size, void* d_ws, size_t ws_size,
                              hipStream_t stream) {
  const float* image   = (const float*)d_in[0];
  // d_in[1]=mask (unused by reference), d_in[2]/[3]=edge indices (grid topology is implicit)
  const float* l1c1_wn = (const float*)d_in[4];
  const float* l1c1_wr = (const float*)d_in[5];
  const float* l1c1_b  = (const float*)d_in[6];
  const float* l1c2_wn = (const float*)d_in[7];
  const float* l1c2_wr = (const float*)d_in[8];
  const float* l1c2_b  = (const float*)d_in[9];
  const float* proj1_w = (const float*)d_in[10];
  const float* proj1_b = (const float*)d_in[11];
  const float* sol1_w  = (const float*)d_in[12];
  const float* sol1_b  = (const float*)d_in[13];
  const float* l2c1_wn = (const float*)d_in[14];
  const float* l2c1_wr = (const float*)d_in[15];
  const float* l2c1_b  = (const float*)d_in[16];
  const float* l2c2_wn = (const float*)d_in[17];
  const float* l2c2_wr = (const float*)d_in[18];
  const float* l2c2_b  = (const float*)d_in[19];
  const float* proj2_w = (const float*)d_in[20];
  const float* proj2_b = (const float*)d_in[21];
  const float* sol2_w  = (const float*)d_in[22];
  const float* sol2_b  = (const float*)d_in[23];

  char* ws = (char*)d_ws;
  _Float16* W1 = (_Float16*)(ws + (0  << 10));   // l1c1 packed: 32x64
  _Float16* W2 = (_Float16*)(ws + (8  << 10));   // l1c2 packed: 128x64
  _Float16* W3 = (_Float16*)(ws + (32 << 10));   // proj1 packed: 64x16
  _Float16* W4 = (_Float16*)(ws + (40 << 10));   // l2c1 packed: 32x64
  _Float16* W5 = (_Float16*)(ws + (48 << 10));   // l2c2 packed: 128x64
  _Float16* W6 = (_Float16*)(ws + (72 << 10));   // proj2 packed: 64x16 (N padded)
  const size_t MB = 1u << 20;
  _Float16* X1h = (_Float16*)(ws + 128 * 1024);             //  2 MB
  _Float16* H1h = (_Float16*)(ws + 128 * 1024 + 2 * MB);    //  8 MB
  _Float16* X2h = (_Float16*)(ws + 128 * 1024 + 10 * MB);   //  4 MB
  _Float16* H3h = (_Float16*)(ws + 128 * 1024 + 14 * MB);   // 32 MB

  float* out = (float*)d_out;
  float* out_sol1 = out;                                    // 65536
  float* out_sol2 = out + 65536;                            // 262144
  float* out_img  = out + 65536 + 262144;                   // 1048576
  float* out_cs   = out + 65536 + 262144 + 1048576;         // 1048576

  auto pk = [&](const float* w0, const float* w1, int F, int Nc, int NT, int KT, _Float16* dst) {
    int total = KT * NT * 512;
    pack_w<<<(total + 255) / 256, 256, 0, stream>>>(w0, w1, F, Nc, NT, KT, total, dst);
  };
  pk(l1c1_wn, l1c1_wr, 16, 64, 4, 1, W1);
  pk(l1c2_wn, l1c2_wr, 64, 64, 4, 4, W2);
  pk(proj1_w, nullptr, 64, 16, 1, 2, W3);
  pk(l2c1_wn, l2c1_wr,  8, 64, 4, 1, W4);
  pk(l2c2_wn, l2c2_wr, 64, 64, 4, 4, W5);
  pk(proj2_w, nullptr, 64,  4, 1, 2, W6);

  patchify1<<<256, 256, 0, stream>>>(image, X1h);
  // level 1: 4*128*(128/16)=4096 wave-tiles, 4 waves/block
  conv_first<16><<<1024, 128, 0, stream>>>(X1h, W1, l1c1_b, H1h, 128);
  conv_second<1><<<1024, 128, 0, stream>>>(H1h, W2, l1c2_b, W3, proj1_b, sol1_w, sol1_b,
                                           out_sol1, X2h, nullptr, nullptr, 128);
  // level 2: 4*256*(256/16)=16384 wave-tiles
  conv_first<8><<<4096, 128, 0, stream>>>(X2h, W4, l2c1_b, H3h, 256);
  conv_second<2><<<4096, 128, 0, stream>>>(H3h, W5, l2c2_b, W6, proj2_b, sol2_w, sol2_b,
                                           out_sol2, nullptr, out_img, out_cs, 256);
}
